// KA_GNN_two_37142877176052
// MI455X (gfx1250) — matrix operand order repaired and checked by reading, not compiled, not run on hardware
//
#include <hip/hip_runtime.h>

// ---------------------------------------------------------------------------
// KA-GNN (KAN polynomial input layer + 3x Fourier-KAN message passing +
// mean-pool + linear readout) for MI455X / gfx1250.
//
// Dense einsums are mapped onto v_wmma_f32_16x16x32_bf16 (fp32 accumulate).
// Fourier features (cos/sin) are generated on the fly into an LDS A-panel
// shared by all 16 waves of a block; coefficients are pre-packed to bf16 in a
// B-fragment-major layout so each lane issues contiguous 128b loads.
// segment_sum uses one wave per edge with float atomics (L2-resident).
// ---------------------------------------------------------------------------

typedef __attribute__((ext_vector_type(16))) __bf16 v16bf;
typedef __attribute__((ext_vector_type(8)))  float  v8f;

#define N_NODES   20000
#define N_EDGES   320000
#define N_GRAPHS  64
#define IN_FEAT   64
#define HIDDEN    256
#define GRIDF     4
#define NLAYERS   3          // Fourier message-passing layers
#define K_LINE    (IN_FEAT * (GRIDF + 1))        // 320
#define K_FOUR    (HIDDEN * GRIDF * 2)           // 2048
#define NEG_SLOPE 0.01f

// ------------------------------ weight packing -----------------------------
// B-fragment layout: Bpack[kt][col][kk], kk = 0..31 (K within tile).
// Lane (col = lane&15) reads 16 contiguous bf16 at kk = (lane>=16)*16.

__global__ void pack_line_kernel(const float* __restrict__ W, __bf16* __restrict__ out) {
    int idx = blockIdx.x * blockDim.x + threadIdx.x;          // 10*256*32
    if (idx >= (K_LINE / 32) * HIDDEN * 32) return;
    int kk  = idx & 31;
    int col = (idx >> 5) & 255;
    int kt  = idx >> 13;
    int f   = kt * 32 + kk;                                   // f < 320 always
    int i   = f / (GRIDF + 1);
    int d   = f - i * (GRIDF + 1);
    float v = W[(col * IN_FEAT + i) * (GRIDF + 1) + d];       // W_line[o][i][d]
    out[idx] = (__bf16)v;
}

__global__ void pack_fourier_kernel(const float* __restrict__ C, __bf16* __restrict__ out) {
    int idx = blockIdx.x * blockDim.x + threadIdx.x;          // 3*64*256*32
    if (idx >= NLAYERS * (K_FOUR / 32) * HIDDEN * 32) return;
    int kk   = idx & 31;
    int col  = (idx >> 5) & 255;
    int kt   = (idx >> 13) & 63;
    int l    = idx >> 19;
    int f    = kt * 32 + kk;           // f = i*8 + trig*4 + g
    int i    = f >> 3;
    int trig = (f >> 2) & 1;
    int g    = f & 3;
    // fourier_coeffs[l][trig][out][in][grid]
    size_t src = ((((size_t)l * 2 + trig) * HIDDEN + col) * HIDDEN + i) * GRIDF + g;
    out[idx] = (__bf16)C[src];
}

// ------------------------------ WMMA helpers -------------------------------

__device__ __forceinline__ v16bf load_a_frag(const __bf16* ap) {
    // lane<16: K = {0..15}; lane>=16: K = {16..31} (abase folded
    // into ap by caller). Two contiguous 16B chunks -> two ds_load_b128.
    v16bf a;
    __builtin_memcpy(&a, ap, 16);
    __builtin_memcpy((char*)&a + 16, ap + 8, 16);
    return a;
}

// ------------------------------ kan_line GEMM ------------------------------
// h0[n, 256] = sum_f powers[n, f] * BL[:, f],  f = i*5 + d, powers = x^d.

__global__ __launch_bounds__(512)
void kan_line_kernel(const float* __restrict__ x, const __bf16* __restrict__ Bpack,
                     float* __restrict__ hout) {
    __shared__ __bf16 Ap[16 * (K_LINE + 8)];                  // pitch 328, 10.5 KB
    const int tid   = threadIdx.x;
    const int node0 = blockIdx.x * 16;

    for (int idx = tid; idx < 16 * K_LINE; idx += 512) {
        int r = idx / K_LINE;
        int f = idx - r * K_LINE;
        int i = f / (GRIDF + 1);
        int d = f - i * (GRIDF + 1);
        float xv = x[(size_t)(node0 + r) * IN_FEAT + i];
        float p = 1.f;
        for (int j = 0; j < d; ++j) p *= xv;
        Ap[r * (K_LINE + 8) + f] = (__bf16)p;
    }
    __syncthreads();

    const int wv    = tid >> 5;
    const int lane  = tid & 31;
    const int col0  = wv * 16;
    const int row   = lane & 15;
    const int abase = (lane >= 16) ? 16 : 0;
    const int kb    = (lane >= 16) ? 16 : 0;

    v8f acc = {0.f, 0.f, 0.f, 0.f, 0.f, 0.f, 0.f, 0.f};
    for (int kt = 0; kt < K_LINE / 32; ++kt) {
        v16bf a = load_a_frag(&Ap[row * (K_LINE + 8) + kt * 32 + abase]);
        v16bf b;
        const __bf16* bp = Bpack + ((size_t)(kt * HIDDEN + col0 + row)) * 32 + kb;
        __builtin_memcpy(&b, bp, 32);
        acc = __builtin_amdgcn_wmma_f32_16x16x32_bf16(false, a, false, b,
                                                      (short)0, acc, false, false);
    }
    const int orow = (lane >> 4) * 8;
    const int ocol = col0 + row;
#pragma unroll
    for (int r = 0; r < 8; ++r)
        hout[(size_t)(node0 + orow + r) * HIDDEN + ocol] = acc[r];
}

// --------------------------- edge scatter (copy_u+sum) ---------------------

__global__ void zero_kernel(float* __restrict__ p, int n) {
    int i = blockIdx.x * blockDim.x + threadIdx.x;
    if (i < n) p[i] = 0.f;
}

__global__ void scatter_kernel(const float* __restrict__ h, const int* __restrict__ src,
                               const int* __restrict__ dst, float* __restrict__ agg,
                               int n_edges) {
    int wave = (int)((blockIdx.x * blockDim.x + threadIdx.x) >> 5);
    int lane = threadIdx.x & 31;
    if (wave >= n_edges) return;
    int s = src[wave], d = dst[wave];
    const float* hr = h   + (size_t)s * HIDDEN;
    float*       ar = agg + (size_t)d * HIDDEN;
#pragma unroll
    for (int i = 0; i < HIDDEN / 32; ++i)
        atomicAdd(&ar[i * 32 + lane], hr[i * 32 + lane]);
}

// --------------------- fused Fourier-KAN GEMM + residual -------------------
// h = leaky_relu( [cos|sin](agg*k) @ coeffs^T + h )

__global__ __launch_bounds__(512)
void fourier_layer_kernel(const float* __restrict__ agg, const __bf16* __restrict__ Bpack,
                          float* __restrict__ h /* in-out */) {
    __shared__ __bf16 Ap[16 * 1032];                          // 33 KB, K chunk = 1024
    const int tid   = threadIdx.x;
    const int node0 = blockIdx.x * 16;
    const int wv    = tid >> 5;
    const int lane  = tid & 31;
    const int col0  = wv * 16;
    const int row   = lane & 15;
    const int abase = (lane >= 16) ? 16 : 0;
    const int kb    = (lane >= 16) ? 16 : 0;

    v8f acc = {0.f, 0.f, 0.f, 0.f, 0.f, 0.f, 0.f, 0.f};

    for (int kc = 0; kc < 2; ++kc) {
        __syncthreads();                                      // protect previous reads
        for (int idx = tid; idx < 16 * 1024; idx += 512) {
            int r  = idx >> 10;
            int fl = idx & 1023;
            int f  = (kc << 10) + fl;                         // f = i*8 + trig*4 + g
            int i    = f >> 3;
            int trig = (f >> 2) & 1;
            int g    = f & 3;
            float av  = agg[(size_t)(node0 + r) * HIDDEN + i];
            float ang = av * (float)(g + 1);
            float v   = trig ? __sinf(ang) : __cosf(ang);
            Ap[r * 1032 + fl] = (__bf16)v;
        }
        __syncthreads();
#pragma unroll 4
        for (int kt = 0; kt < 32; ++kt) {
            v16bf a = load_a_frag(&Ap[row * 1032 + kt * 32 + abase]);
            v16bf b;
            int ktg = (kc << 5) + kt;
            const __bf16* bp = Bpack + ((size_t)(ktg * HIDDEN + col0 + row)) * 32 + kb;
            __builtin_memcpy(&b, bp, 32);
            acc = __builtin_amdgcn_wmma_f32_16x16x32_bf16(false, a, false, b,
                                                          (short)0, acc, false, false);
        }
    }

    const int orow = (lane >> 4) * 8;
    const int ocol = col0 + row;
#pragma unroll
    for (int r = 0; r < 8; ++r) {
        size_t off = (size_t)(node0 + orow + r) * HIDDEN + ocol;
        float hv = acc[r] + h[off];
        h[off] = hv > 0.f ? hv : NEG_SLOPE * hv;
    }
}

// ------------------------------ pool + readout -----------------------------

__global__ void pool_kernel(const float* __restrict__ h, const int* __restrict__ gid,
                            float* __restrict__ sums, float* __restrict__ counts,
                            int n_nodes) {
    int wave = (int)((blockIdx.x * blockDim.x + threadIdx.x) >> 5);
    int lane = threadIdx.x & 31;
    if (wave >= n_nodes) return;
    int g = gid[wave];
    const float* hr = h    + (size_t)wave * HIDDEN;
    float*       sr = sums + (size_t)g * HIDDEN;
#pragma unroll
    for (int i = 0; i < HIDDEN / 32; ++i)
        atomicAdd(&sr[i * 32 + lane], hr[i * 32 + lane]);
    if (lane == 0) atomicAdd(&counts[g], 1.0f);
}

__global__ void readout_kernel(const float* __restrict__ sums, const float* __restrict__ counts,
                               const float* __restrict__ Wout, const float* __restrict__ bout,
                               float* __restrict__ out) {
    __shared__ float red[HIDDEN];
    int g = blockIdx.x;
    int c = threadIdx.x;
    float cnt = counts[g];
    cnt = cnt > 1.f ? cnt : 1.f;
    float y = sums[(size_t)g * HIDDEN + c] / cnt;
    red[c] = Wout[c * 2 + 0] + Wout[c * 2 + 1] * y;           // degree-1 KAN poly
    __syncthreads();
    for (int s = HIDDEN / 2; s > 0; s >>= 1) {
        if (c < s) red[c] += red[c + s];
        __syncthreads();
    }
    if (c == 0) out[g] = 1.f / (1.f + __expf(-(red[0] + bout[0])));
}

// ------------------------------- launcher ----------------------------------

extern "C" void kernel_launch(void* const* d_in, const int* in_sizes, int n_in,
                              void* d_out, int out_size, void* d_ws, size_t ws_size,
                              hipStream_t stream) {
    const float* h_in  = (const float*)d_in[0];
    const int*   src   = (const int*)d_in[1];
    const int*   dst   = (const int*)d_in[2];
    const int*   gid   = (const int*)d_in[3];
    const float* Wline = (const float*)d_in[4];
    const float* Fcoef = (const float*)d_in[5];
    const float* Wout  = (const float*)d_in[6];
    const float* bout  = (const float*)d_in[7];
    float* out = (float*)d_out;

    char*  ws  = (char*)d_ws;
    size_t off = 0;
    auto alloc = [&](size_t bytes) -> void* {
        void* p = ws + off;
        off += (bytes + 255) & ~(size_t)255;
        return p;
    };
    float*  hbuf   = (float*)alloc((size_t)N_NODES * HIDDEN * sizeof(float));   // 20.5 MB
    float*  agg    = (float*)alloc((size_t)N_NODES * HIDDEN * sizeof(float));   // 20.5 MB
    __bf16* BL     = (__bf16*)alloc((size_t)(K_LINE / 32) * HIDDEN * 32 * 2);   // 160 KB
    __bf16* BF     = (__bf16*)alloc((size_t)NLAYERS * (K_FOUR / 32) * HIDDEN * 32 * 2); // 3 MB
    float*  sums   = (float*)alloc((size_t)N_GRAPHS * HIDDEN * sizeof(float));
    float*  counts = (float*)alloc((size_t)N_GRAPHS * sizeof(float));
    (void)in_sizes; (void)n_in; (void)out_size; (void)ws_size;

    // 1) pack weights to bf16 B-fragment layout
    {
        int n = (K_LINE / 32) * HIDDEN * 32;
        pack_line_kernel<<<(n + 255) / 256, 256, 0, stream>>>(Wline, BL);
        n = NLAYERS * (K_FOUR / 32) * HIDDEN * 32;
        pack_fourier_kernel<<<(n + 255) / 256, 256, 0, stream>>>(Fcoef, BF);
    }

    // 2) input polynomial-KAN layer (WMMA GEMM, 20000 x 320 x 256)
    kan_line_kernel<<<N_NODES / 16, 512, 0, stream>>>(h_in, BL, hbuf);

    // 3) 3x message passing: scatter-add then fused Fourier GEMM + residual
    for (int l = 0; l < NLAYERS; ++l) {
        int n = N_NODES * HIDDEN;
        zero_kernel<<<(n + 255) / 256, 256, 0, stream>>>(agg, n);
        scatter_kernel<<<(N_EDGES * 32) / 256, 256, 0, stream>>>(hbuf, src, dst, agg, N_EDGES);
        fourier_layer_kernel<<<N_NODES / 16, 512, 0, stream>>>(
            agg, BF + (size_t)l * (K_FOUR / 32) * HIDDEN * 32, hbuf);
    }

    // 4) mean pool per graph + readout
    {
        int n = N_GRAPHS * HIDDEN;
        zero_kernel<<<(n + 255) / 256, 256, 0, stream>>>(sums, n);
        zero_kernel<<<1, N_GRAPHS, 0, stream>>>(counts, N_GRAPHS);
        pool_kernel<<<(N_NODES * 32 + 255) / 256, 256, 0, stream>>>(hbuf, gid, sums, counts, N_NODES);
        readout_kernel<<<N_GRAPHS, HIDDEN, 0, stream>>>(sums, counts, Wout, bout, out);
    }
}
